// MoeMlp_49924699848936
// MI455X (gfx1250) — matrix operand order repaired and checked by reading, not compiled, not run on hardware
//
#include <hip/hip_runtime.h>
#include <hip/hip_bf16.h>
#include <math.h>

// Problem constants (B=4, S=2048 -> T=8192)
#define T_TOK 8192
#define DDIM  1024
#define FDIM  2048
#define NEXP  8
#define BM    32     // tokens per expert tile (two 16-row M tiles share B frags)
#define FC    128    // F chunk per iteration (8 waves x 16 cols)
#define NWAVE 8      // 256 threads / wave32

typedef __attribute__((ext_vector_type(16))) __bf16        v16bf;
typedef __attribute__((ext_vector_type(8)))  float         v8f;
typedef __attribute__((ext_vector_type(8)))  unsigned int  v8ui;

// ---- fp32 -> bf16 round-to-nearest-even (prepack, fallback B, x gather, H store) ----
__device__ __forceinline__ unsigned short f2bf(float f) {
  unsigned int u = __builtin_bit_cast(unsigned int, f);
  u = (u + 0x7FFFu + ((u >> 16) & 1u)) >> 16;
  return (unsigned short)u;
}
__device__ __forceinline__ unsigned int pack2(float a, float b) {
  return (unsigned int)f2bf(a) | ((unsigned int)f2bf(b) << 16);
}

// ---- WMMA wrapper: D = A(16x32 bf16) x B(32x16 bf16) + C(f32) ----
__device__ __forceinline__ v8f wmma_bf16(v16bf a, v16bf b, v8f c) {
  return __builtin_amdgcn_wmma_f32_16x16x32_bf16(
      false, a, false, b, (short)0, c, false, false);
}

// ---- A fragment (16x32 bf16) from LDS, contiguous k = 16*half + s (same as B) ----
__device__ __forceinline__ v16bf load_A_lds(const unsigned short* base,
                                            int row_stride, int m0, int k0) {
  const int lane = threadIdx.x & 31;
  const int m    = m0 + (lane & 15);
  const int hh   = lane >> 4;
  v8ui p;
#pragma unroll
  for (int v = 0; v < 8; ++v) {
    int kp = 2 * v + (hh ? 16 : 0);
    p[v] = *(const unsigned int*)(base + m * row_stride + k0 + kp);
  }
  return __builtin_bit_cast(v16bf, p);
}

// ---- B fragment from PRE-SWIZZLED packed bf16 weights: one 32B per-lane load
// Packed layout (uints): ((tk * ntn + tn) * 32 + lane) * 8 + v
__device__ __forceinline__ v16bf load_B_packed(const unsigned int* __restrict__ P,
                                               int ntn, int tk, int tn) {
  const int lane = threadIdx.x & 31;
  const v8ui p = *(const v8ui*)(P + (((size_t)tk * ntn + tn) * 32 + lane) * 8);
  return __builtin_bit_cast(v16bf, p);
}

// ---- B fragment converted on the fly from fp32 (fallback when ws too small)
__device__ __forceinline__ v16bf load_B_global(const float* __restrict__ W,
                                               int ldw, int k0, int n0) {
  const int lane  = threadIdx.x & 31;
  const int n     = n0 + (lane & 15);
  const int khalf = lane >> 4;
  v8ui p;
#pragma unroll
  for (int v = 0; v < 8; ++v) {
    int k = k0 + khalf * 16 + 2 * v;
    p[v] = pack2(W[(size_t)k * ldw + n], W[(size_t)(k + 1) * ldw + n]);
  }
  return __builtin_bit_cast(v16bf, p);
}

// ============  kernel P: prepack fp32 weights -> swizzled bf16 fragments  ====
__global__ void moe_prepack_kernel(const float* __restrict__ src,
                                   unsigned int* __restrict__ dst,
                                   int K, int N, int nmat) {
  const size_t per = (size_t)K * N / 2;       // uints per matrix
  const size_t total = per * nmat;
  size_t j = (size_t)blockIdx.x * blockDim.x + threadIdx.x;
  const size_t stride = (size_t)gridDim.x * blockDim.x;
  const int ntn = N >> 4;
  for (; j < total; j += stride) {
    int e = (int)(j / per);
    size_t r = j % per;
    int kp = (int)(r / N);        // row pair
    int n  = (int)(r % N);
    int k  = kp << 1;
    int tk    = k >> 5;
    int khalf = (k >> 4) & 1;
    int v     = (k >> 1) & 7;
    int tn    = n >> 4;
    int lane  = (khalf << 4) | (n & 15);
    const float* s = src + (size_t)e * K * N;
    unsigned int w = pack2(s[(size_t)k * N + n], s[(size_t)(k + 1) * N + n]);
    dst[(size_t)e * per + (((size_t)tk * ntn + tn) * 32 + lane) * 8 + v] = w;
  }
}

// =====================  kernel 0: zero output + counters  ====================
__global__ void moe_zero_kernel(float* __restrict__ out, size_t n,
                                int* __restrict__ cnt) {
  if (blockIdx.x == 0 && threadIdx.x < NEXP) cnt[threadIdx.x] = 0;
  size_t i = (size_t)blockIdx.x * blockDim.x + threadIdx.x;
  size_t stride = (size_t)gridDim.x * blockDim.x;
  for (; i < n; i += stride) out[i] = 0.0f;
}

// =====================  kernel 1: router (wave per token)  ===================
__global__ void moe_router_kernel(const float* __restrict__ x,
                                  const float* __restrict__ gate_w,
                                  const float* __restrict__ gate_b,
                                  float* __restrict__ logits_out,
                                  int* __restrict__ cnt,
                                  int* __restrict__ btok,
                                  float* __restrict__ bwgt) {
  const int wave = threadIdx.x >> 5;
  const int lane = threadIdx.x & 31;
  const int t = blockIdx.x * (blockDim.x >> 5) + wave;
  if (t >= T_TOK) return;

  float acc[NEXP];
#pragma unroll
  for (int e = 0; e < NEXP; ++e) acc[e] = 0.0f;

  const float* xr = x + (size_t)t * DDIM;
  for (int d = lane; d < DDIM; d += 32) {
    float xv = xr[d];
    const float* gw = gate_w + (size_t)d * NEXP;
#pragma unroll
    for (int e = 0; e < NEXP; ++e) acc[e] = fmaf(xv, gw[e], acc[e]);
  }
#pragma unroll
  for (int off = 16; off > 0; off >>= 1) {
#pragma unroll
    for (int e = 0; e < NEXP; ++e) acc[e] += __shfl_xor(acc[e], off, 32);
  }
  if (lane != 0) return;

  float lg[NEXP], p[NEXP];
  float m = -1e30f;
#pragma unroll
  for (int e = 0; e < NEXP; ++e) { lg[e] = acc[e] + gate_b[e]; m = fmaxf(m, lg[e]); }
  float s = 0.0f;
#pragma unroll
  for (int e = 0; e < NEXP; ++e) { p[e] = __expf(lg[e] - m); s += p[e]; }
  float inv = 1.0f / s;
#pragma unroll
  for (int e = 0; e < NEXP; ++e) {
    p[e] *= inv;
    logits_out[(size_t)t * NEXP + e] = lg[e];
  }
  int i0 = 0; float v0 = p[0];
#pragma unroll
  for (int e = 1; e < NEXP; ++e) if (p[e] > v0) { v0 = p[e]; i0 = e; }
  int i1 = -1; float v1 = -1.0f;
#pragma unroll
  for (int e = 0; e < NEXP; ++e) {
    if (e == i0) continue;
    if (p[e] > v1) { v1 = p[e]; i1 = e; }
  }
  float wsum = v0 + v1;
  float w0 = v0 / wsum, w1 = v1 / wsum;
  int pos0 = atomicAdd(&cnt[i0], 1);
  btok[i0 * T_TOK + pos0] = t;  bwgt[i0 * T_TOK + pos0] = w0;
  int pos1 = atomicAdd(&cnt[i1], 1);
  btok[i1 * T_TOK + pos1] = t;  bwgt[i1 * T_TOK + pos1] = w1;
}

// ============  kernel 2: fused expert FFN (gate/up/silu/down)  ===============
template <bool PACKED>
__global__ void __launch_bounds__(NWAVE * 32, 1)
moe_expert_kernel(const float* __restrict__ x,
                  const float* __restrict__ Wg,
                  const float* __restrict__ bg,
                  const float* __restrict__ Wu,
                  const float* __restrict__ bu,
                  const float* __restrict__ Wd,
                  const float* __restrict__ bd,
                  const unsigned int* __restrict__ pWg,
                  const unsigned int* __restrict__ pWu,
                  const unsigned int* __restrict__ pWd,
                  const int* __restrict__ cnt,
                  const int* __restrict__ btok,
                  const float* __restrict__ bwgt,
                  float* __restrict__ out) {
  const int tiles_per_e = T_TOK / BM;
  const int e    = blockIdx.x / tiles_per_e;
  const int tile = blockIdx.x % tiles_per_e;
  const int n_e  = cnt[e];
  const int row0 = tile * BM;
  if (row0 >= n_e) return;   // uniform across block

  extern __shared__ unsigned char smem[];
  unsigned short* Xs   = (unsigned short*)smem;                       // BM x DDIM bf16
  unsigned short* Hs   = (unsigned short*)(smem + BM * DDIM * 2);     // BM x FC bf16
  int*            tokS = (int*)(smem + BM * DDIM * 2 + BM * FC * 2);  // BM
  float*          wgtS = (float*)(tokS + BM);                         // BM

  const int tid = threadIdx.x;
  if (tid < BM) {
    int r = row0 + tid;
    int t = (r < n_e) ? btok[e * T_TOK + r] : -1;
    tokS[tid] = t;
    wgtS[tid] = (r < n_e) ? bwgt[e * T_TOK + r] : 0.0f;
  }
  __syncthreads();

  for (int i = tid; i < BM * DDIM; i += blockDim.x) {
    int r = i >> 10;            // / DDIM
    int d = i & (DDIM - 1);
    int t = tokS[r];
    float v = (t >= 0) ? x[(size_t)t * DDIM + d] : 0.0f;
    Xs[i] = f2bf(v);
  }
  __syncthreads();

  const int wave = tid >> 5;
  const int lane = tid & 31;
  const int col  = lane & 15;
  const int hh   = lane >> 4;

  const float* Wge = Wg + (size_t)e * DDIM * FDIM;
  const float* Wue = Wu + (size_t)e * DDIM * FDIM;
  const float* Wde = Wd + (size_t)e * FDIM * DDIM;
  const unsigned int* pWge = pWg + (size_t)e * (DDIM * FDIM / 2);
  const unsigned int* pWue = pWu + (size_t)e * (DDIM * FDIM / 2);
  const unsigned int* pWde = pWd + (size_t)e * (FDIM * DDIM / 2);

  const int NTN_F = FDIM / 16;   // 128
  const int NTN_D = DDIM / 16;   // 64
  const int dbase = wave * (DDIM / NWAVE);

  const v8f z8 = {0.f, 0.f, 0.f, 0.f, 0.f, 0.f, 0.f, 0.f};
  v8f accD[2][8];
#pragma unroll
  for (int m2 = 0; m2 < 2; ++m2)
#pragma unroll
    for (int i = 0; i < 8; ++i) accD[m2][i] = z8;

  for (int f0 = 0; f0 < FDIM; f0 += FC) {
    const int fn = f0 + wave * 16;
    const int tnF = fn >> 4;
    v8f g0 = z8, g1 = z8, u0 = z8, u1 = z8;
#pragma unroll 2
    for (int k0 = 0; k0 < DDIM; k0 += 32) {
      v16bf a0 = load_A_lds(Xs, DDIM, 0, k0);
      v16bf a1 = load_A_lds(Xs, DDIM, 16, k0);
      v16bf bg_, bu_;
      if (PACKED) {
        const int tk = k0 >> 5;
        bg_ = load_B_packed(pWge, NTN_F, tk, tnF);
        bu_ = load_B_packed(pWue, NTN_F, tk, tnF);
      } else {
        bg_ = load_B_global(Wge, FDIM, k0, fn);
        bu_ = load_B_global(Wue, FDIM, k0, fn);
      }
      g0 = wmma_bf16(a0, bg_, g0);
      g1 = wmma_bf16(a1, bg_, g1);
      u0 = wmma_bf16(a0, bu_, u0);
      u1 = wmma_bf16(a1, bu_, u1);
    }
    float bgv = bg[(size_t)e * FDIM + fn + col];
    float buv = bu[(size_t)e * FDIM + fn + col];
#pragma unroll
    for (int r = 0; r < 8; ++r) {
      int rowm = r + 8 * hh;                  // C layout: M = r + 8*half
      float gv0 = g0[r] + bgv, uv0 = u0[r] + buv;
      float gv1 = g1[r] + bgv, uv1 = u1[r] + buv;
      float s0 = gv0 / (1.0f + __expf(-gv0));
      float s1 = gv1 / (1.0f + __expf(-gv1));
      Hs[rowm * FC + wave * 16 + col]        = f2bf(s0 * uv0);
      Hs[(rowm + 16) * FC + wave * 16 + col] = f2bf(s1 * uv1);
    }
    __syncthreads();

#pragma unroll 1
    for (int kk = 0; kk < FC; kk += 32) {
      v16bf a0 = load_A_lds(Hs, FC, 0, kk);
      v16bf a1 = load_A_lds(Hs, FC, 16, kk);
      const int tkD = (f0 + kk) >> 5;
#pragma unroll
      for (int n = 0; n < 8; ++n) {
        v16bf b;
        if (PACKED) {
          b = load_B_packed(pWde, NTN_D, tkD, (dbase >> 4) + n);
        } else {
          b = load_B_global(Wde + (size_t)f0 * DDIM, DDIM, kk, dbase + n * 16);
        }
        accD[0][n] = wmma_bf16(a0, b, accD[0][n]);
        accD[1][n] = wmma_bf16(a1, b, accD[1][n]);
      }
    }
    __syncthreads();
  }

#pragma unroll
  for (int m2 = 0; m2 < 2; ++m2) {
#pragma unroll
    for (int n = 0; n < 8; ++n) {
      int d = dbase + n * 16 + col;
      float bdv = bd[(size_t)e * DDIM + d];
#pragma unroll
      for (int r = 0; r < 8; ++r) {
        int rowm = m2 * 16 + r + 8 * hh;
        int t = tokS[rowm];
        if (t >= 0) {
          float w = wgtS[rowm];
          atomicAdd(out + (size_t)t * DDIM + d, w * (accD[m2][n][r] + bdv));
        }
      }
    }
  }
}

// ============================  launcher  =====================================
extern "C" void kernel_launch(void* const* d_in, const int* in_sizes, int n_in,
                              void* d_out, int out_size, void* d_ws,
                              size_t ws_size, hipStream_t stream) {
  (void)in_sizes; (void)n_in; (void)out_size;
  const float* x      = (const float*)d_in[0];
  const float* gate_w = (const float*)d_in[1];
  const float* gate_b = (const float*)d_in[2];
  const float* Wg     = (const float*)d_in[3];
  const float* bg     = (const float*)d_in[4];
  const float* Wu     = (const float*)d_in[5];
  const float* bu     = (const float*)d_in[6];
  const float* Wd     = (const float*)d_in[7];
  const float* bd     = (const float*)d_in[8];

  float* out    = (float*)d_out;
  float* logits = out + (size_t)T_TOK * DDIM;

  unsigned char* ws = (unsigned char*)d_ws;
  size_t off = 0;
  int*   cnt  = (int*)(ws + off);  off += 64;
  int*   btok = (int*)(ws + off);  off += (size_t)NEXP * T_TOK * 4;
  float* bwgt = (float*)(ws + off); off += (size_t)NEXP * T_TOK * 4;
  off = (off + 255) & ~(size_t)255;

  const size_t packUints = (size_t)NEXP * DDIM * FDIM / 2;   // per weight tensor
  unsigned int* pWg = (unsigned int*)(ws + off);
  unsigned int* pWu = pWg + packUints;
  unsigned int* pWd = pWu + packUints;
  const size_t need = off + 3 * packUints * 4;
  const bool packed = (ws_size >= need);

  moe_zero_kernel<<<2048, 256, 0, stream>>>(out, (size_t)T_TOK * DDIM, cnt);

  if (packed) {
    moe_prepack_kernel<<<8192, 256, 0, stream>>>(Wg, pWg, DDIM, FDIM, NEXP);
    moe_prepack_kernel<<<8192, 256, 0, stream>>>(Wu, pWu, DDIM, FDIM, NEXP);
    moe_prepack_kernel<<<8192, 256, 0, stream>>>(Wd, pWd, FDIM, DDIM, NEXP);
  }

  moe_router_kernel<<<T_TOK / (256 / 32), 256, 0, stream>>>(
      x, gate_w, gate_b, logits, cnt, btok, bwgt);

  size_t ldsBytes = (size_t)BM * DDIM * 2 + (size_t)BM * FC * 2 +
                    (size_t)BM * 4 + (size_t)BM * 4;   // ~72.5 KB
  dim3 grid(NEXP * (T_TOK / BM));
  dim3 block(NWAVE * 32);
  if (packed) {
    moe_expert_kernel<true><<<grid, block, ldsBytes, stream>>>(
        x, Wg, bg, Wu, bu, Wd, bd, pWg, pWu, pWd, cnt, btok, bwgt, out);
  } else {
    moe_expert_kernel<false><<<grid, block, ldsBytes, stream>>>(
        x, Wg, bg, Wu, bu, Wd, bd, pWg, pWu, pWd, cnt, btok, bwgt, out);
  }
}
